// Decoder_2783138807985
// MI455X (gfx1250) — compile-verified, matrix operand order checked by reading, not run on hardware
//
#include <hip/hip_runtime.h>
#include <math.h>

// ---------------- problem constants (from reference) ----------------
static constexpr int NV = 10000;   // vocab
static constexpr int NE = 512;     // embed
static constexpr int NH = 512;     // lstm hidden
static constexpr int ND = 2048;    // encoder dim
static constexpr int NA = 512;     // attention dim
static constexpr int NB = 64;      // batch
static constexpr int NG = 14;      // enc image size
static constexpr int NP = NG * NG; // 196 pixels
static constexpr int NTCAP = 20;   // caption length
static constexpr int NT = NTCAP - 1; // 19 decode steps

// ---------------- WMMA vector types ----------------
typedef __attribute__((ext_vector_type(16))) __bf16 v16bf;
typedef __attribute__((ext_vector_type(8)))  float  v8f;

// Load two 8-float chunks (K=[off..off+8) and [off+16..off+24)) and convert to
// a bf16 WMMA fragment. Caller already applied off = k + (lane>>4)*8.
__device__ __forceinline__ v16bf load_cvt_frag(const float* __restrict__ p) {
    const float4 a0 = *reinterpret_cast<const float4*>(p);
    const float4 a1 = *reinterpret_cast<const float4*>(p + 4);
    const float4 a2 = *reinterpret_cast<const float4*>(p + 16);
    const float4 a3 = *reinterpret_cast<const float4*>(p + 20);
    v16bf f;
    f[0]  = (__bf16)a0.x; f[1]  = (__bf16)a0.y; f[2]  = (__bf16)a0.z; f[3]  = (__bf16)a0.w;
    f[4]  = (__bf16)a1.x; f[5]  = (__bf16)a1.y; f[6]  = (__bf16)a1.z; f[7]  = (__bf16)a1.w;
    f[8]  = (__bf16)a2.x; f[9]  = (__bf16)a2.y; f[10] = (__bf16)a2.z; f[11] = (__bf16)a2.w;
    f[12] = (__bf16)a3.x; f[13] = (__bf16)a3.y; f[14] = (__bf16)a3.z; f[15] = (__bf16)a3.w;
    return f;
}

// -------- generic WMMA GEMM: C = [beta*C] + A[M,K] @ W[N,K]^T (+bias) --------
// Register-blocked: each wave computes an (MT*16) x (NW*16) strip of C.
// Requires M % (16*MT) == 0, K % 32 == 0; N handled per-16-tile (N%16==0).
// 4 waves per block; grid = (ceil(Ntiles_n / (4*NW)), M/(16*MT)).
// Wave id goes through readfirstlane so all tile guards are SCALAR branches:
// EXEC stays all-1s around every WMMA (ISA requirement) and the K-loop carries
// no saveexec masking.
// If lens != nullptr: row index is a batch index; store 0 when !(t < lens[row]-1).
template <int MT, int NW>
__global__ __launch_bounds__(128)
void k_gemm(const float* __restrict__ A, const float* __restrict__ W,
            const float* __restrict__ bias, float* __restrict__ C,
            int M, int N, int K, long long ldC, int beta,
            const int* __restrict__ lens, int t)
{
    const int lane   = threadIdx.x & 31;
    const int wave   = __builtin_amdgcn_readfirstlane(threadIdx.x >> 5); // SGPR
    const int nTile0 = (blockIdx.x * 4 + wave) * NW;   // units of 16 columns (scalar)
    const int mTile0 = blockIdx.y * MT;                // units of 16 rows (scalar)
    if (nTile0 * 16 >= N) return;   // scalar branch

    const int lr = lane & 15;
    const int hi = lane >> 4;       // which K half-chunk this lane covers

    const float* arow[MT];
#pragma unroll
    for (int mi = 0; mi < MT; ++mi)
        arow[mi] = A + (size_t)((mTile0 + mi) * 16 + lr) * K + hi * 8;
    const float* wrow[NW];
#pragma unroll
    for (int j = 0; j < NW; ++j)
        wrow[j] = W + (size_t)((nTile0 + j) * 16 + lr) * K + hi * 8;

    v8f acc[MT][NW];
#pragma unroll
    for (int mi = 0; mi < MT; ++mi)
#pragma unroll
        for (int j = 0; j < NW; ++j)
            acc[mi][j] = v8f{0.f, 0.f, 0.f, 0.f, 0.f, 0.f, 0.f, 0.f};

    for (int k = 0; k < K; k += 32) {
        v16bf af[MT];
#pragma unroll
        for (int mi = 0; mi < MT; ++mi) af[mi] = load_cvt_frag(arow[mi] + k);
#pragma unroll
        for (int j = 0; j < NW; ++j) {
            if ((nTile0 + j) * 16 < N) {                 // scalar guard
                __builtin_prefetch(wrow[j] + k + 128, 0, 1); // stream weights
                v16bf wf = load_cvt_frag(wrow[j] + k);
#pragma unroll
                for (int mi = 0; mi < MT; ++mi)
                    acc[mi][j] = __builtin_amdgcn_wmma_f32_16x16x32_bf16(
                        false, af[mi], false, wf, (short)0, acc[mi][j], false, false);
            }
        }
    }

#pragma unroll
    for (int mi = 0; mi < MT; ++mi) {
#pragma unroll
        for (int j = 0; j < NW; ++j) {
            if ((nTile0 + j) * 16 < N) {
                const int col  = (nTile0 + j) * 16 + lr;
                const float bc = bias ? bias[col] : 0.f;
#pragma unroll
                for (int r = 0; r < 8; ++r) {
                    const int row = (mTile0 + mi) * 16 + r + hi * 8;
                    float* outp = C + (size_t)row * ldC + col;
                    float v = acc[mi][j][r] + bc;
                    if (beta) v += *outp;
                    if (lens && !(t < lens[row] - 1)) v = 0.f;
                    *outp = v;
                }
            }
        }
    }
}

// ---------------- mean over pixels: meanf[b,d] = mean_p feats[b,p,d] ----------------
__global__ __launch_bounds__(256)
void k_mean(const float* __restrict__ feats, float* __restrict__ meanf)
{
    int idx = blockIdx.x * 256 + threadIdx.x;          // over NB*ND
    if (idx >= NB * ND) return;
    int b = idx / ND, d = idx - b * ND;
    const float* p = feats + (size_t)b * NP * ND + d;
    float s = 0.f;
    for (int q = 0; q < NP; ++q) s += p[(size_t)q * ND];
    meanf[idx] = s * (1.0f / NP);
}

// ---------------- fused attention: e -> softmax -> alpha (out) -> awe ----------------
// one block (256 thr) per batch element
__global__ __launch_bounds__(256)
void k_attention(const float* __restrict__ att1, const float* __restrict__ att2,
                 const float* __restrict__ watt, const float* __restrict__ batt,
                 const float* __restrict__ feats, const int* __restrict__ lens, int t,
                 float* __restrict__ alphas_out, float* __restrict__ awe)
{
    __shared__ float s_e[256];
    __shared__ float s_att2[NA];
    __shared__ float s_watt[NA];
    __shared__ float s_red[8];

    const int b    = blockIdx.x;
    const int tid  = threadIdx.x;
    const int lane = tid & 31;
    const int wave = tid >> 5;

    for (int a = tid; a < NA; a += 256) {
        s_att2[a] = att2[b * NA + a];
        s_watt[a] = watt[a];
    }
    __syncthreads();

    // e[p] = relu(att1[b,p,:] + att2[b,:]) . watt + batt
    const float bt = batt[0];
    for (int p = wave; p < NP; p += 8) {
        const float* row = att1 + ((size_t)b * NP + p) * NA;
        float acc = 0.f;
        for (int a = lane; a < NA; a += 32) {
            float v = row[a] + s_att2[a];
            v = v > 0.f ? v : 0.f;
            acc += v * s_watt[a];
        }
        for (int off = 16; off > 0; off >>= 1) acc += __shfl_down(acc, off, 32);
        if (lane == 0) s_e[p] = acc + bt;
    }
    __syncthreads();

    // softmax over NP entries
    float xv = (tid < NP) ? s_e[tid] : -3.4e38f;
    float m = xv;
    for (int off = 16; off > 0; off >>= 1) m = fmaxf(m, __shfl_down(m, off, 32));
    if (lane == 0) s_red[wave] = m;
    __syncthreads();
    if (tid == 0) {
        float mm = s_red[0];
        for (int i = 1; i < 8; ++i) mm = fmaxf(mm, s_red[i]);
        s_red[0] = mm;
    }
    __syncthreads();
    const float mx = s_red[0];
    const float ex = (tid < NP) ? __expf(xv - mx) : 0.f;
    __syncthreads();
    float sv = ex;
    for (int off = 16; off > 0; off >>= 1) sv += __shfl_down(sv, off, 32);
    if (lane == 0) s_red[wave] = sv;
    __syncthreads();
    if (tid == 0) {
        float tt = 0.f;
        for (int i = 0; i < 8; ++i) tt += s_red[i];
        s_red[0] = 1.0f / tt;
    }
    __syncthreads();
    const float alpha = ex * s_red[0];
    const bool actv = (t < lens[b] - 1);
    if (tid < NP) {
        s_e[tid] = alpha;
        alphas_out[(size_t)b * NT * NP + (size_t)t * NP + tid] = actv ? alpha : 0.f;
    }
    __syncthreads();

    // awe[b,d] = sum_p alpha[p] * feats[b,p,d]  (coalesced in d)
    const float* fb = feats + (size_t)b * NP * ND;
    for (int d = tid; d < ND; d += 256) {
        float s = 0.f;
        for (int p = 0; p < NP; ++p) s += s_e[p] * fb[(size_t)p * ND + d];
        awe[b * ND + d] = s;
    }
}

// ---------------- build LSTM input x = [emb[cap[b,t]], sigmoid(gatepre)*awe] ----------------
__global__ __launch_bounds__(256)
void k_build_x(const float* __restrict__ emb, const int* __restrict__ captions, int t,
               const float* __restrict__ gatepre, const float* __restrict__ awe,
               float* __restrict__ x)
{
    int idx = blockIdx.x * 256 + threadIdx.x;          // over NB*(NE+ND)
    if (idx >= NB * (NE + ND)) return;
    int b = idx / (NE + ND), j = idx - b * (NE + ND);
    float v;
    if (j < NE) {
        int tok = captions[b * NTCAP + t];
        v = emb[(size_t)tok * NE + j];
    } else {
        int d = j - NE;
        float g = 1.0f / (1.0f + __expf(-gatepre[b * ND + d]));
        v = g * awe[b * ND + d];
    }
    x[idx] = v;
}

// ---------------- LSTM pointwise + masked state update ----------------
__global__ __launch_bounds__(256)
void k_lstm(const float* __restrict__ gates, const int* __restrict__ lens, int t,
            float* __restrict__ h, float* __restrict__ c, float* __restrict__ hnew)
{
    int idx = blockIdx.x * 256 + threadIdx.x;          // over NB*NH
    if (idx >= NB * NH) return;
    int b = idx / NH, u = idx - b * NH;
    const float* g = gates + (size_t)b * 4 * NH;
    float gi = g[u], gf = g[NH + u], gg = g[2 * NH + u], go = g[3 * NH + u];
    float si = 1.0f / (1.0f + __expf(-gi));
    float sf = 1.0f / (1.0f + __expf(-gf));
    float so = 1.0f / (1.0f + __expf(-go));
    float cn = sf * c[idx] + si * tanhf(gg);
    float hn = so * tanhf(cn);
    hnew[idx] = hn;
    if (t < lens[b] - 1) { h[idx] = hn; c[idx] = cn; }
}

// ---------------- host-side orchestration ----------------
extern "C" void kernel_launch(void* const* d_in, const int* in_sizes, int n_in,
                              void* d_out, int out_size, void* d_ws, size_t ws_size,
                              hipStream_t stream) {
    (void)in_sizes; (void)n_in; (void)out_size; (void)ws_size;

    const float* features = (const float*)d_in[0];   // [B,14,14,D] == [B*P, D]
    const int*   captions = (const int*)d_in[1];     // [B, 20]
    const int*   lengths  = (const int*)d_in[2];     // [B]
    const float* emb   = (const float*)d_in[3];
    const float* W_ih  = (const float*)d_in[4];  const float* b_ih = (const float*)d_in[5];
    const float* W_hh  = (const float*)d_in[6];  const float* b_hh = (const float*)d_in[7];
    const float* Wh    = (const float*)d_in[8];  const float* bh   = (const float*)d_in[9];
    const float* Wc    = (const float*)d_in[10]; const float* bc   = (const float*)d_in[11];
    const float* Wb    = (const float*)d_in[12]; const float* bb   = (const float*)d_in[13];
    const float* Wcls  = (const float*)d_in[14]; const float* bcls = (const float*)d_in[15];
    const float* Wenc  = (const float*)d_in[16]; const float* benc = (const float*)d_in[17];
    const float* Wdec  = (const float*)d_in[18]; const float* bdec = (const float*)d_in[19];
    const float* Watt  = (const float*)d_in[20]; const float* batt = (const float*)d_in[21];

    float* y_out = (float*)d_out;                          // [B,T,V]
    float* a_out = y_out + (size_t)NB * NT * NV;           // [B,T,P]

    // workspace carve-up (~29 MB of f32)
    float* w = (float*)d_ws;
    float* att1    = w; w += (size_t)NB * NP * NA;         // [B*P, A]
    float* att2    = w; w += (size_t)NB * NA;
    float* hbuf    = w; w += (size_t)NB * NH;
    float* cbuf    = w; w += (size_t)NB * NH;
    float* hnew    = w; w += (size_t)NB * NH;
    float* meanf   = w; w += (size_t)NB * ND;
    float* awe     = w; w += (size_t)NB * ND;
    float* gatepre = w; w += (size_t)NB * ND;
    float* xbuf    = w; w += (size_t)NB * (NE + ND);
    float* gates   = w; w += (size_t)NB * 4 * NH;

    // big: 32x64 strip per wave (8 WMMA chains); small: 16x16 tile per wave
    auto gemm_big = [&](const float* Ap, const float* Wp, const float* biasp, float* Cp,
                        int M, int N, int K, long long ldC, int beta,
                        const int* lens, int t) {
        dim3 grid((N / 16 + 4 * 4 - 1) / (4 * 4), M / 32);
        k_gemm<2, 4><<<grid, dim3(128), 0, stream>>>(Ap, Wp, biasp, Cp, M, N, K, ldC,
                                                     beta, lens, t);
    };
    auto gemm_small = [&](const float* Ap, const float* Wp, const float* biasp, float* Cp,
                          int M, int N, int K, long long ldC, int beta,
                          const int* lens, int t) {
        dim3 grid((N / 16 + 3) / 4, M / 16);
        k_gemm<1, 1><<<grid, dim3(128), 0, stream>>>(Ap, Wp, biasp, Cp, M, N, K, ldC,
                                                     beta, lens, t);
    };

    // ---- precompute ----
    k_mean<<<dim3((NB * ND + 255) / 256), dim3(256), 0, stream>>>(features, meanf);
    gemm_small(meanf, Wh, bh, hbuf, NB, NH, ND, NH, 0, nullptr, 0);            // h0
    gemm_small(meanf, Wc, bc, cbuf, NB, NH, ND, NH, 0, nullptr, 0);            // c0
    gemm_big(features, Wenc, benc, att1, NB * NP, NA, ND, NA, 0, nullptr, 0);  // att1

    // ---- decode loop (static T) ----
    for (int t = 0; t < NT; ++t) {
        gemm_small(hbuf, Wdec, bdec, att2, NB, NA, NH, NA, 0, nullptr, 0);     // att2
        k_attention<<<dim3(NB), dim3(256), 0, stream>>>(
            att1, att2, Watt, batt, features, lengths, t, a_out, awe);
        gemm_small(hbuf, Wb, bb, gatepre, NB, ND, NH, ND, 0, nullptr, 0);      // gate pre
        k_build_x<<<dim3((NB * (NE + ND) + 255) / 256), dim3(256), 0, stream>>>(
            emb, captions, t, gatepre, awe, xbuf);
        gemm_small(xbuf, W_ih, b_ih, gates, NB, 4 * NH, NE + ND, 4 * NH, 0, nullptr, 0);
        gemm_small(hbuf, W_hh, b_hh, gates, NB, 4 * NH, NH, 4 * NH, 1, nullptr, 0);
        k_lstm<<<dim3((NB * NH + 255) / 256), dim3(256), 0, stream>>>(
            gates, lengths, t, hbuf, cbuf, hnew);
        // logits: masked rows -> 0, stored straight into y_out[:, t, :]
        gemm_big(hnew, Wcls, bcls, y_out + (size_t)t * NV, NB, NV, NH,
                 (long long)NT * NV, 0, lengths, t);
    }
}
